// VectorQuantizer_35837207117904
// MI455X (gfx1250) — compile-verified, matrix operand order checked by reading, not run on hardware
//
#include <hip/hip_runtime.h>
#include <hip/hip_bf16.h>

typedef _Float16 v16h __attribute__((ext_vector_type(16)));
typedef _Float16 v8h  __attribute__((ext_vector_type(8)));
typedef float    v8f  __attribute__((ext_vector_type(8)));
typedef float    v4f  __attribute__((ext_vector_type(4)));

#define NUM_CODES   512
#define DIM         64
#define T_LEN       4096
#define BATCH       32
#define N_COLS      (BATCH * T_LEN)           // 131072 row-vectors
#define ROW_PITCH_H 72                        // LDS pitch per E row, in halves (144 B, 16B aligned)
#define LDS_EK2_OFF (NUM_CODES * ROW_PITCH_H * 2)   // 73728 bytes
#define LDS_BYTES   (LDS_EK2_OFF + NUM_CODES * 4)   // + 2 KB Ek2
#define LOSS_SCALE  (1.25 / 8388608.0)        // 1.25 / (B*D*T)

static __device__ __forceinline__ v8f wmma_f16(v16h a, v16h b, v8f c) {
    return __builtin_amdgcn_wmma_f32_16x16x32_f16(false, a, false, b, (short)0, c,
                                                  false, false);
}

__global__ void vq_zero(double* acc) { *acc = 0.0; }

__global__ void vq_finalize(const double* __restrict__ acc, float* __restrict__ out) {
    out[0] = (float)(*acc * LOSS_SCALE);
}

__global__ __launch_bounds__(256) void vq_main(const float* __restrict__ z,
                                               const float* __restrict__ E,
                                               float* __restrict__ out,
                                               double* __restrict__ lossAcc) {
    extern __shared__ __align__(16) char smem[];
    _Float16* sE   = (_Float16*)smem;                    // (-512*E) f16, pitch 72 halves
    float*    sEk2 = (float*)(smem + LDS_EK2_OFF);       // 256 * ||E_k||^2

    const int tid   = threadIdx.x;
    const int lane  = tid & 31;
    const int wave  = tid >> 5;
    const int sub   = lane & 15;                         // column / row within 16-tile
    const int half8 = (lane & 16) ? 8 : 0;               // M/K half select per WMMA layout

    // ---- stage codebook into LDS: f16(-512*E) + scaled squared norms ----
    for (int r = tid; r < NUM_CODES; r += 256) {
        const v4f* Er = (const v4f*)(E + r * DIM);
        _Float16* dst = sE + r * ROW_PITCH_H;
        float ss = 0.f;
        #pragma unroll
        for (int i = 0; i < DIM / 4; ++i) {
            v4f v = Er[i];
            #pragma unroll
            for (int j = 0; j < 4; ++j) {
                float e = v[j];
                ss += e * e;
                dst[i * 4 + j] = (_Float16)(e * -512.0f);
            }
        }
        sEk2[r] = ss * 256.0f;
    }
    __syncthreads();

    // ---- build resident B operands: z columns, split hi/lo f16 ----
    const int n0 = blockIdx.x * 256 + wave * 32;         // 32 columns per wave
    v16h b_hi[2][2], b_lo[2][2];                         // [ntile][kchunk]
    #pragma unroll
    for (int nt = 0; nt < 2; ++nt) {
        int n = n0 + nt * 16 + sub;
        int b = n >> 12, t = n & (T_LEN - 1);
        const float* zc = z + (size_t)b * DIM * T_LEN + t;
        #pragma unroll
        for (int c = 0; c < 2; ++c) {
            v16h hi, lo;
            #pragma unroll
            for (int e = 0; e < 16; ++e) {
                int d = c * 32 + half8 + ((e < 8) ? e : e + 8);   // WMMA K mapping
                float zv = zc[(size_t)d * T_LEN];
                _Float16 h = (_Float16)zv;
                hi[e] = h;
                lo[e] = (_Float16)(zv - (float)h);
            }
            b_hi[nt][c] = hi;
            b_lo[nt][c] = lo;
        }
    }

    // ---- sweep all 512 codes: dist = 256*(||E_k||^2 - 2 z.E_k), track argmin ----
    float minv0 = 3.402823466e38f, minv1 = 3.402823466e38f;
    int   mini0 = 0, mini1 = 0;
    const int rowsub = sub * ROW_PITCH_H;

    for (int mt = 0; mt < 32; ++mt) {
        const v4f* pk = (const v4f*)(sEk2 + mt * 16 + half8);
        v4f ka = pk[0], kb = pk[1];
        v8f c0 = {ka[0], ka[1], ka[2], ka[3], kb[0], kb[1], kb[2], kb[3]};
        v8f c1 = c0;

        const _Float16* arow = sE + mt * 16 * ROW_PITCH_H + rowsub;
        v16h a[2];
        #pragma unroll
        for (int c = 0; c < 2; ++c) {
            v8h p0 = *(const v8h*)(arow + c * 32 + half8);
            v8h p1 = *(const v8h*)(arow + c * 32 + 16 + half8);
            v16h av;
            #pragma unroll
            for (int e = 0; e < 8; ++e) { av[e] = p0[e]; av[e + 8] = p1[e]; }
            a[c] = av;
        }

        c0 = wmma_f16(a[0], b_hi[0][0], c0);
        c0 = wmma_f16(a[1], b_hi[0][1], c0);
        c0 = wmma_f16(a[0], b_lo[0][0], c0);
        c0 = wmma_f16(a[1], b_lo[0][1], c0);
        c1 = wmma_f16(a[0], b_hi[1][0], c1);
        c1 = wmma_f16(a[1], b_hi[1][1], c1);
        c1 = wmma_f16(a[0], b_lo[1][0], c1);
        c1 = wmma_f16(a[1], b_lo[1][1], c1);

        const int kbase = mt * 16 + half8;
        #pragma unroll
        for (int j = 0; j < 8; ++j) {
            float v0 = c0[j];
            if (v0 < minv0) { minv0 = v0; mini0 = kbase + j; }
            float v1 = c1[j];
            if (v1 < minv1) { minv1 = v1; mini1 = kbase + j; }
        }
    }

    // combine lane l with l^16 (same column, other M half); tie -> lower index
    {
        float ov = __shfl_xor(minv0, 16, 32); int oi = __shfl_xor(mini0, 16, 32);
        if (ov < minv0 || (ov == minv0 && oi < mini0)) { minv0 = ov; mini0 = oi; }
        ov = __shfl_xor(minv1, 16, 32); oi = __shfl_xor(mini1, 16, 32);
        if (ov < minv1 || (ov == minv1 && oi < mini1)) { minv1 = ov; mini1 = oi; }
    }
    const int code = (lane < 16) ? mini0 : mini1;        // lane l owns column n0+l

    // ---- gather winning code, write q in [B,D,T], accumulate loss ----
    const int n = n0 + lane;
    const int b = n >> 12, t = n & (T_LEN - 1);
    const float* zc = z + (size_t)b * DIM * T_LEN + t;
    float*       qc = out + 1 + (size_t)b * DIM * T_LEN + t;
    const float* Er = E + (size_t)code * DIM;
    float acc = 0.f;
    #pragma unroll 8
    for (int d = 0; d < DIM; ++d) {
        float e  = Er[d];
        float zv = zc[(size_t)d * T_LEN];
        float df = e - zv;
        acc += df * df;
        qc[(size_t)d * T_LEN] = e;
    }
    out[1 + (size_t)N_COLS * DIM + n] = (float)code;     // idx output

    #pragma unroll
    for (int s = 16; s > 0; s >>= 1) acc += __shfl_xor(acc, s, 32);
    if (lane == 0) atomicAdd(lossAcc, (double)acc);
}

extern "C" void kernel_launch(void* const* d_in, const int* in_sizes, int n_in,
                              void* d_out, int out_size, void* d_ws, size_t ws_size,
                              hipStream_t stream) {
    const float* z = (const float*)d_in[0];
    const float* E = (const float*)d_in[1];
    float* out = (float*)d_out;
    double* acc = (double*)d_ws;

    vq_zero<<<dim3(1), dim3(1), 0, stream>>>(acc);
    vq_main<<<dim3(N_COLS / 256), dim3(256), LDS_BYTES, stream>>>(z, E, out, acc);
    vq_finalize<<<dim3(1), dim3(1), 0, stream>>>(acc, out);
}
